// HybridGPT_48558900248924
// MI455X (gfx1250) — matrix operand order repaired and checked by reading, not compiled, not run on hardware
//
#include <hip/hip_runtime.h>
#include <hip/hip_bf16.h>
#include <math.h>

// ---------------------------------------------------------------------------
// Problem constants (from the reference)
// ---------------------------------------------------------------------------
#define B_   2
#define T_   1024
#define D_   1024
#define H_   16
#define HKV_ 4
#define HD_  64
#define HG_  8
#define DK_  64
#define DV_  128

typedef __attribute__((ext_vector_type(16))) __bf16 v16bf;
typedef __attribute__((ext_vector_type(8)))  __bf16 v8bf;
typedef __attribute__((ext_vector_type(8)))  float  v8f;

// ---------------------------------------------------------------------------
// bf16 helpers (RNE convert, no arithmetic on __bf16 itself)
// ---------------------------------------------------------------------------
__device__ __forceinline__ __bf16 f2bf(float f) {
  unsigned u = __builtin_bit_cast(unsigned, f);
  u += 0x7FFFu + ((u >> 16) & 1u);
  unsigned short h = (unsigned short)(u >> 16);
  return __builtin_bit_cast(__bf16, h);
}
__device__ __forceinline__ float bf2f(__bf16 b) {
  unsigned short h = __builtin_bit_cast(unsigned short, b);
  unsigned u = ((unsigned)h) << 16;
  return __builtin_bit_cast(float, u);
}

// Load 16 bf16 as two 16-byte chunks (p0, p1 both 16B aligned)
__device__ __forceinline__ v16bf ld16(const __bf16* p0, const __bf16* p1) {
  uint4 lo = *(const uint4*)p0;
  uint4 hi = *(const uint4*)p1;
  v8bf a = __builtin_bit_cast(v8bf, lo);
  v8bf b = __builtin_bit_cast(v8bf, hi);
  return __builtin_shufflevector(a, b, 0,1,2,3,4,5,6,7,8,9,10,11,12,13,14,15);
}

#define WMMA_BF16(a, b, c) \
  __builtin_amdgcn_wmma_f32_16x16x32_bf16(false, (a), false, (b), (short)0, (c), false, false)

// ---------------------------------------------------------------------------
// fp32 -> bf16 conversion
// ---------------------------------------------------------------------------
__global__ void k_f32_to_bf16(const float* __restrict__ src, __bf16* __restrict__ dst, int n) {
  int i = blockIdx.x * blockDim.x + threadIdx.x;
  if (i < n) dst[i] = f2bf(src[i]);
}

// ---------------------------------------------------------------------------
// C[M,N] = A[M,K] @ W[N,K]^T   (bf16 in, f32 out).
// Register-blocked: one 32(M) x 64(N) tile per wave = 2 A-operands x
// 4 B-operands -> 8 WMMAs per 32-wide K chunk (~21 FLOP/B of operand traffic,
// A reused 4x / B reused 2x from VGPRs).
// A-operand lane layout (16-bit 16x32): lane<16 -> rows 0..15, K=k0..k0+15;
// lane>=16 -> K=k0+16..k0+31 (same contiguous k order as B).
// B-operand: lane<16 -> N=lane, K=k0..k0+15; lane>=16 -> K=k0+16..k0+31.
// C layout: VGPR v -> row v (+8 for upper half), col = lane&15.
// ---------------------------------------------------------------------------
__global__ void k_gemm_bf16(const __bf16* __restrict__ A, const __bf16* __restrict__ W,
                            float* __restrict__ C, int M, int N, int K) {
  const int lane = threadIdx.x & 31;
  const int wave = threadIdx.x >> 5;
  const int tile = blockIdx.x * 4 + wave;
  const int tilesN = N >> 6;                 // 64-wide N tiles
  const int tm = tile / tilesN;              // 32-wide M tiles
  const int tn = tile % tilesN;
  if (tm >= (M >> 5)) return;
  const int half = lane >> 4;
  const int l15  = lane & 15;

  const long arow0 = (long)(tm * 32 + l15) * K;        // A rows 0..15
  const long arow1 = (long)(tm * 32 + 16 + l15) * K;   // A rows 16..31
  const long brow0 = (long)(tn * 64 + 0  + l15) * K;   // B col blocks
  const long brow1 = (long)(tn * 64 + 16 + l15) * K;
  const long brow2 = (long)(tn * 64 + 32 + l15) * K;
  const long brow3 = (long)(tn * 64 + 48 + l15) * K;

  v8f acc00 = {}, acc01 = {}, acc02 = {}, acc03 = {};
  v8f acc10 = {}, acc11 = {}, acc12 = {}, acc13 = {};

  for (int k0 = 0; k0 < K; k0 += 32) {
    const int aoff = k0 + half * 16;
    const int boff = k0 + half * 16;
    const __bf16* ap0 = A + arow0 + aoff;
    const __bf16* ap1 = A + arow1 + aoff;
    const __bf16* bp0 = W + brow0 + boff;
    const __bf16* bp1 = W + brow1 + boff;
    const __bf16* bp2 = W + brow2 + boff;
    const __bf16* bp3 = W + brow3 + boff;

    v16bf a0 = ld16(ap0, ap0 + 8);
    v16bf a1 = ld16(ap1, ap1 + 8);
    v16bf b0 = ld16(bp0, bp0 + 8);
    v16bf b1 = ld16(bp1, bp1 + 8);
    v16bf b2 = ld16(bp2, bp2 + 8);
    v16bf b3 = ld16(bp3, bp3 + 8);

    __builtin_prefetch(ap0 + 64, 0, 1);
    __builtin_prefetch(ap1 + 64, 0, 1);
    __builtin_prefetch(bp0 + 64, 0, 1);
    __builtin_prefetch(bp2 + 64, 0, 1);

    acc00 = WMMA_BF16(a0, b0, acc00);
    acc01 = WMMA_BF16(a0, b1, acc01);
    acc02 = WMMA_BF16(a0, b2, acc02);
    acc03 = WMMA_BF16(a0, b3, acc03);
    acc10 = WMMA_BF16(a1, b0, acc10);
    acc11 = WMMA_BF16(a1, b1, acc11);
    acc12 = WMMA_BF16(a1, b2, acc12);
    acc13 = WMMA_BF16(a1, b3, acc13);
  }

  const int crow0 = tm * 32 + half * 8;
  const int crow1 = crow0 + 16;
  const int ccol  = tn * 64 + l15;
#pragma unroll
  for (int v = 0; v < 8; ++v) {
    float* r0 = C + (long)(crow0 + v) * N + ccol;
    float* r1 = C + (long)(crow1 + v) * N + ccol;
    r0[0]  = acc00[v];
    r0[16] = acc01[v];
    r0[32] = acc02[v];
    r0[48] = acc03[v];
    r1[0]  = acc10[v];
    r1[16] = acc11[v];
    r1[32] = acc12[v];
    r1[48] = acc13[v];
  }
}

// ---------------------------------------------------------------------------
// Post-process projections: per-head RMS norm, RoPE, q_gain.  One wave per
// (b, t, head-task).  Lane d owns the RoPE pair (d, d+32).  V is stored
// transposed: vt[b,hkv,d,T] so P*V B-operands are per-lane contiguous.
// ---------------------------------------------------------------------------
__global__ void k_postproc(const float* __restrict__ q_raw, const float* __restrict__ k_raw,
                           const float* __restrict__ v_raw, const float* __restrict__ q_gain,
                           __bf16* __restrict__ q_bf, __bf16* __restrict__ k_bf,
                           __bf16* __restrict__ vt_bf) {
  const int lane = threadIdx.x & 31;
  const int gw   = blockIdx.x * (blockDim.x >> 5) + (threadIdx.x >> 5);
  const int id24 = gw % 24;
  const int bt   = gw / 24;
  if (bt >= B_ * T_) return;
  const int b = bt / T_, t = bt % T_;

  if (id24 < H_) {                 // ---- Q heads: rms + rope + gain
    const int h = id24;
    const float* src = q_raw + (long)bt * D_ + h * HD_;
    float x1 = src[lane], x2 = src[lane + 32];
    float ss = x1 * x1 + x2 * x2;
#pragma unroll
    for (int off = 16; off; off >>= 1) ss += __shfl_xor(ss, off, 32);
    float r = rsqrtf(ss * (1.0f / 64.0f) + 1e-6f);
    x1 *= r; x2 *= r;
    float fr = (float)t * __expf(-((2.0f * lane) / 64.0f) * __logf(10000.0f));
    float c = __cosf(fr), s = __sinf(fr);
    float g = q_gain[h];
    __bf16* dst = q_bf + ((long)(b * H_ + h) * T_ + t) * HD_;
    dst[lane]      = f2bf((x1 * c + x2 * s) * g);
    dst[lane + 32] = f2bf((-x1 * s + x2 * c) * g);
  } else if (id24 < H_ + HKV_) {   // ---- K heads: rms + rope
    const int h = id24 - H_;
    const float* src = k_raw + (long)bt * (HKV_ * HD_) + h * HD_;
    float x1 = src[lane], x2 = src[lane + 32];
    float ss = x1 * x1 + x2 * x2;
#pragma unroll
    for (int off = 16; off; off >>= 1) ss += __shfl_xor(ss, off, 32);
    float r = rsqrtf(ss * (1.0f / 64.0f) + 1e-6f);
    x1 *= r; x2 *= r;
    float fr = (float)t * __expf(-((2.0f * lane) / 64.0f) * __logf(10000.0f));
    float c = __cosf(fr), s = __sinf(fr);
    __bf16* dst = k_bf + ((long)(b * HKV_ + h) * T_ + t) * HD_;
    dst[lane]      = f2bf(x1 * c + x2 * s);
    dst[lane + 32] = f2bf(-x1 * s + x2 * c);
  } else {                         // ---- V heads: transpose store
    const int h = id24 - H_ - HKV_;
    const float* src = v_raw + (long)bt * (HKV_ * HD_) + h * HD_;
    __bf16* dst = vt_bf + (long)(b * HKV_ + h) * HD_ * T_;
    dst[(long)lane * T_ + t]        = f2bf(src[lane]);
    dst[(long)(lane + 32) * T_ + t] = f2bf(src[lane + 32]);
  }
}

// ---------------------------------------------------------------------------
// Flash attention: one wave per (b, h, 16-row q tile).  32-kpos tiles:
// 4 WMMAs for S = Q K^T (two 16x16 score blocks, d accumulated over 2x32),
// online softmax with shfl within 16-lane halves (matches C striping),
// P staged bf16 through LDS, 4 WMMAs for O += P V over 4 chunks of 16 vdims.
// ---------------------------------------------------------------------------
__global__ void k_attn(const __bf16* __restrict__ qb, const __bf16* __restrict__ kb,
                       const __bf16* __restrict__ vtb, __bf16* __restrict__ ob) {
  __shared__ __bf16 pbuf[4][16][32];     // per-wave P slab
  const int lane = threadIdx.x & 31;
  const int wv   = threadIdx.x >> 5;
  const int gw   = blockIdx.x * 4 + wv;
  const int qt   = gw % (T_ / 16);
  const int bh   = gw / (T_ / 16);
  if (bh >= B_ * H_) return;
  const int b = bh / H_, h = bh % H_, hk = h / (H_ / HKV_);
  const int half = lane >> 4, l15 = lane & 15;

  // Q operand (A-matrix, 16x64 split into two d-windows of 32)
  const __bf16* qrow = qb + ((long)bh * T_ + qt * 16 + l15) * HD_;
  const __bf16* qa0  = qrow + 0  + half * 16;
  const __bf16* qa1  = qrow + 32 + half * 16;
  v16bf aq0 = ld16(qa0, qa0 + 8);
  v16bf aq1 = ld16(qa1, qa1 + 8);

  float mrow[8], lrow[8];
  v8f o0 = {}, o1 = {}, o2 = {}, o3 = {};
#pragma unroll
  for (int v = 0; v < 8; ++v) { mrow[v] = -1e30f; lrow[v] = 0.0f; }

  const int qmax = qt * 16 + 15;
  const __bf16* kbase = kb  + (long)(b * HKV_ + hk) * T_  * HD_;
  const __bf16* vbase = vtb + (long)(b * HKV_ + hk) * HD_ * T_;
  const float scale = 0.125f;   // 1/sqrt(64)

  for (int j = 0; j * 32 <= qmax; ++j) {
    // ---- scores: two 16x16 blocks over kpos [j*32, j*32+32)
    v8f s0 = {}, s1 = {};
    {
      const __bf16* kp0 = kbase + (long)(j * 32 + l15) * HD_;
      const __bf16* kp1 = kbase + (long)(j * 32 + 16 + l15) * HD_;
      v16bf bk;
      bk = ld16(kp0 + 0  + half * 16, kp0 + 0  + half * 16 + 8);
      s0 = WMMA_BF16(aq0, bk, s0);
      bk = ld16(kp0 + 32 + half * 16, kp0 + 32 + half * 16 + 8);
      s0 = WMMA_BF16(aq1, bk, s0);
      bk = ld16(kp1 + 0  + half * 16, kp1 + 0  + half * 16 + 8);
      s1 = WMMA_BF16(aq0, bk, s1);
      bk = ld16(kp1 + 32 + half * 16, kp1 + 32 + half * 16 + 8);
      s1 = WMMA_BF16(aq1, bk, s1);
    }

    // ---- causal mask + online softmax (row stats per lane's 8 rows)
#pragma unroll
    for (int v = 0; v < 8; ++v) {
      const int qpos = qt * 16 + v + half * 8;
      const int kp0 = j * 32 + l15, kp1 = kp0 + 16;
      float x0 = (kp0 <= qpos) ? s0[v] * scale : -1e30f;
      float x1 = (kp1 <= qpos) ? s1[v] * scale : -1e30f;
      float mloc = fmaxf(x0, x1);
      mloc = fmaxf(mloc, __shfl_xor(mloc, 1, 32));
      mloc = fmaxf(mloc, __shfl_xor(mloc, 2, 32));
      mloc = fmaxf(mloc, __shfl_xor(mloc, 4, 32));
      mloc = fmaxf(mloc, __shfl_xor(mloc, 8, 32));
      float mnew = fmaxf(mrow[v], mloc);
      float corr = __expf(mrow[v] - mnew);
      float p0 = __expf(x0 - mnew);
      float p1 = __expf(x1 - mnew);
      pbuf[wv][v + half * 8][l15]      = f2bf(p0);
      pbuf[wv][v + half * 8][16 + l15] = f2bf(p1);
      float ls = p0 + p1;
      ls += __shfl_xor(ls, 1, 32);
      ls += __shfl_xor(ls, 2, 32);
      ls += __shfl_xor(ls, 4, 32);
      ls += __shfl_xor(ls, 8, 32);
      lrow[v] = lrow[v] * corr + ls;
      mrow[v] = mnew;
      o0[v] *= corr; o1[v] *= corr; o2[v] *= corr; o3[v] *= corr;
    }

    // cross-lane LDS store -> load hazard within the wave
    asm volatile("s_wait_dscnt 0" ::: "memory");

    // ---- P as A-operand (16x32 over local kpos window)
    v16bf ap = {};
    {
      const int r   = l15;
      const int kb8 = half * 16;
#pragma unroll
      for (int e = 0; e < 8; ++e)  ap[e] = pbuf[wv][r][kb8 + e];
#pragma unroll
      for (int e = 8; e < 16; ++e) ap[e] = pbuf[wv][r][kb8 + e];
    }

    // ---- O += P * V over 4 chunks of 16 vdims (Vt is [d][T] contiguous)
    {
      const long col = (long)(j * 32 + half * 16);
      const __bf16* vp;
      v16bf bv;
      vp = vbase + (long)(0 * 16 + l15) * T_ + col;  bv = ld16(vp, vp + 8);
      o0 = WMMA_BF16(ap, bv, o0);
      vp = vbase + (long)(1 * 16 + l15) * T_ + col;  bv = ld16(vp, vp + 8);
      o1 = WMMA_BF16(ap, bv, o1);
      vp = vbase + (long)(2 * 16 + l15) * T_ + col;  bv = ld16(vp, vp + 8);
      o2 = WMMA_BF16(ap, bv, o2);
      vp = vbase + (long)(3 * 16 + l15) * T_ + col;  bv = ld16(vp, vp + 8);
      o3 = WMMA_BF16(ap, bv, o3);
    }
  }

  // ---- finalize: divide by softmax sum, write bf16 rows for the Wo GEMM
#pragma unroll
  for (int v = 0; v < 8; ++v) {
    float inv = 1.0f / lrow[v];
    long row = (long)(b * T_ + qt * 16 + v + half * 8);
    __bf16* d = ob + row * D_ + h * HD_;
    d[0 * 16 + l15] = f2bf(o0[v] * inv);
    d[1 * 16 + l15] = f2bf(o1[v] * inv);
    d[2 * 16 + l15] = f2bf(o2[v] * inv);
    d[3 * 16 + l15] = f2bf(o3[v] * inv);
  }
}

// ---------------------------------------------------------------------------
// Gated delta-net recurrence.  One block per (b, head-group); thread v owns
// state column S[:, v] (64 f32 in registers).  Serial over T; k/q l2-norms
// computed by waves 0/1 into LDS per step.  Accumulates into d_out.
// ---------------------------------------------------------------------------
__global__ void k_gdn(const float* __restrict__ gq, const float* __restrict__ gk,
                      const float* __restrict__ gv, const float* __restrict__ alpha,
                      const float* __restrict__ beta, float* __restrict__ out) {
  __shared__ float ksh[DK_], qsh[DK_], ab[2];
  const int tid  = threadIdx.x;          // 0..127
  const int lane = tid & 31, wv = tid >> 5;
  const int b = blockIdx.x / HG_, h = blockIdx.x % HG_;

  float S[DK_];
#pragma unroll
  for (int d = 0; d < DK_; ++d) S[d] = 0.0f;

  for (int t = 0; t < T_; ++t) {
    const long base = (long)(b * T_ + t) * HG_ + h;
    if (wv == 0) {
      const float* kp = gk + base * DK_;
      float x1 = kp[lane], x2 = kp[lane + 32];
      float ss = x1 * x1 + x2 * x2;
#pragma unroll
      for (int off = 16; off; off >>= 1) ss += __shfl_xor(ss, off, 32);
      float inv = 1.0f / fmaxf(sqrtf(ss), 1e-6f);
      ksh[lane] = x1 * inv; ksh[lane + 32] = x2 * inv;
      if (lane == 0) { ab[0] = alpha[base]; ab[1] = beta[base]; }
    } else if (wv == 1) {
      const float* qp = gq + base * DK_;
      float x1 = qp[lane], x2 = qp[lane + 32];
      float ss = x1 * x1 + x2 * x2;
#pragma unroll
      for (int off = 16; off; off >>= 1) ss += __shfl_xor(ss, off, 32);
      float inv = 1.0f / fmaxf(sqrtf(ss), 1e-6f);
      qsh[lane] = x1 * inv; qsh[lane + 32] = x2 * inv;
    }
    __syncthreads();

    const float a   = ab[0];
    const float bta = ab[1];
    const float vt  = gv[base * DV_ + tid];

    float kS = 0.0f;
#pragma unroll
    for (int d = 0; d < DK_; ++d) kS += ksh[d] * S[d];
    float o = 0.0f;
#pragma unroll
    for (int d = 0; d < DK_; ++d) {
      float s = a * (S[d] - bta * ksh[d] * kS) + ksh[d] * vt;
      S[d] = s;
      o += s * qsh[d];
    }
    out[(long)(b * T_ + t) * D_ + h * DV_ + tid] += o;
    __syncthreads();
  }
}

// ---------------------------------------------------------------------------
// Host side
// ---------------------------------------------------------------------------
extern "C" void kernel_launch(void* const* d_in, const int* in_sizes, int n_in,
                              void* d_out, int out_size, void* d_ws, size_t ws_size,
                              hipStream_t stream) {
  (void)in_sizes; (void)n_in; (void)out_size; (void)ws_size;
  const float* x     = (const float*)d_in[0];
  const float* Wq    = (const float*)d_in[1];
  const float* Wk    = (const float*)d_in[2];
  const float* Wv    = (const float*)d_in[3];
  const float* Wo    = (const float*)d_in[4];
  const float* qgain = (const float*)d_in[5];
  const float* gq    = (const float*)d_in[6];
  const float* gk    = (const float*)d_in[7];
  const float* gv    = (const float*)d_in[8];
  const float* alpha = (const float*)d_in[9];
  const float* beta  = (const float*)d_in[10];
  float* out = (float*)d_out;

  char* ws = (char*)d_ws;
  size_t off = 0;
  auto wsalloc = [&](size_t bytes) -> char* {
    char* p = ws + off;
    off = (off + bytes + 255) & ~(size_t)255;
    return p;
  };

  const int MT = B_ * T_;                       // 2048 rows
  __bf16* x_bf    = (__bf16*)wsalloc((size_t)MT * D_ * 2);
  __bf16* wq_bf   = (__bf16*)wsalloc((size_t)D_ * D_ * 2);
  __bf16* wk_bf   = (__bf16*)wsalloc((size_t)(HKV_*HD_) * D_ * 2);
  __bf16* wv_bf   = (__bf16*)wsalloc((size_t)(HKV_*HD_) * D_ * 2);
  __bf16* wo_bf   = (__bf16*)wsalloc((size_t)D_ * D_ * 2);
  float*  q_raw   = (float*)wsalloc((size_t)MT * D_ * 4);
  float*  k_raw   = (float*)wsalloc((size_t)MT * (HKV_*HD_) * 4);
  float*  v_raw   = (float*)wsalloc((size_t)MT * (HKV_*HD_) * 4);
  __bf16* q_bf    = (__bf16*)wsalloc((size_t)B_ * H_ * T_ * HD_ * 2);
  __bf16* k_bf    = (__bf16*)wsalloc((size_t)B_ * HKV_ * T_ * HD_ * 2);
  __bf16* vt_bf   = (__bf16*)wsalloc((size_t)B_ * HKV_ * HD_ * T_ * 2);
  __bf16* attn_bf = (__bf16*)wsalloc((size_t)MT * D_ * 2);

  // 1) bf16 conversions
  auto cvt = [&](const float* s, __bf16* d, int n) {
    k_f32_to_bf16<<<(n + 255) / 256, 256, 0, stream>>>(s, d, n);
  };
  cvt(x,  x_bf,  MT * D_);
  cvt(Wq, wq_bf, D_ * D_);
  cvt(Wk, wk_bf, HKV_ * HD_ * D_);
  cvt(Wv, wv_bf, HKV_ * HD_ * D_);
  cvt(Wo, wo_bf, D_ * D_);

  // 2) QKV projections (WMMA GEMMs): tiles = (M/32)*(N/64), 4 waves/block
  k_gemm_bf16<<<(MT / 32) * (D_ / 64) / 4, 128, 0, stream>>>(x_bf, wq_bf, q_raw, MT, D_, D_);
  k_gemm_bf16<<<(MT / 32) * ((HKV_*HD_) / 64) / 4, 128, 0, stream>>>(x_bf, wk_bf, k_raw, MT, HKV_*HD_, D_);
  k_gemm_bf16<<<(MT / 32) * ((HKV_*HD_) / 64) / 4, 128, 0, stream>>>(x_bf, wv_bf, v_raw, MT, HKV_*HD_, D_);

  // 3) RMS + RoPE + gain, V transpose.  24 wave-tasks per (b,t), 8 waves/block
  k_postproc<<<MT * 24 / 8, 256, 0, stream>>>(q_raw, k_raw, v_raw, qgain, q_bf, k_bf, vt_bf);

  // 4) Flash attention: one wave per 16-row q tile -> B*H*(T/16) waves
  k_attn<<<B_ * H_ * (T_ / 16) / 4, 128, 0, stream>>>(q_bf, k_bf, vt_bf, attn_bf);

  // 5) Output projection writes y_attn into d_out
  k_gemm_bf16<<<(MT / 32) * (D_ / 64) / 4, 128, 0, stream>>>(attn_bf, wo_bf, out, MT, D_, D_);

  // 6) Gated delta-net recurrence accumulates into d_out
  k_gdn<<<B_ * HG_, 128, 0, stream>>>(gq, gk, gv, alpha, beta, out);
}
